// GNN_10574209482835
// MI455X (gfx1250) — matrix operand order repaired and checked by reading, not compiled, not run on hardware
//
#include <hip/hip_runtime.h>

// ---------------------------------------------------------------------------
// CDNA5 / gfx1250 GAT (2-layer) pipeline.
// GEMMs via v_wmma_f32_16x16x32_bf16 (wave32, 16x64 block per wave, software-
// pipelined global loads; sched_barrier pins loads ahead of the WMMA chain so
// the per-WMMA waits are partial and loads stay in flight across the MMAs).
// Softmax-aggregation via L2-resident f32 atomics (working set << 192MB L2).
// ---------------------------------------------------------------------------

typedef __attribute__((ext_vector_type(16))) __bf16 v16bf;
typedef __attribute__((ext_vector_type(8)))  __bf16 v8bf;
typedef __attribute__((ext_vector_type(8)))  float  v8f;

#define NNODES   10000
#define F_IN     768
#define HEADS1   8
#define CH1      128
#define DIM1     1024     /* HEADS1*CH1 */
#define DIM2     768
#define NEGSLOPE 0.2f
#define EPSF     1e-16f

// ---- workspace layout (bytes, all 256-aligned) ----------------------------
#define OFF_XB   ((size_t)0)                          // x bf16        15,360,000
#define OFF_W1B  ((size_t)15360000)                   // W1 bf16        1,572,864
#define OFF_W2B  ((size_t)16932864)                   // W2 bf16        1,572,864
#define OFF_H1   ((size_t)18505728)                   // h1 f32 (later h2 bf16) 40,960,000
#define OFF_ATT  ((size_t)59465728)                   // aS,aD,max,denom  1,280,000
#define OFF_EX   ((size_t)60745728)                   // per-edge e/exp   5,120,000
#define OFF_AGG  ((size_t)65865728)                   // agg1 f32 (later h2 f32) 40,960,000
// total: 106,825,728 bytes

__device__ __forceinline__ unsigned short f2bf(float f) {
    unsigned u = __float_as_uint(f);
    unsigned r = u + 0x7fffu + ((u >> 16) & 1u);   // round-to-nearest-even
    return (unsigned short)(r >> 16);
}
__device__ __forceinline__ int fmap(float f) {     // monotone float->int map
    int i = __float_as_int(f);
    return i >= 0 ? i : (i ^ 0x7fffffff);
}
__device__ __forceinline__ float funmap(int i) {   // involution inverse
    return __int_as_float(i >= 0 ? i : (i ^ 0x7fffffff));
}

// ---------------------------------------------------------------------------
__global__ void k_f2bf(const float* __restrict__ in,
                       unsigned short* __restrict__ out, int n) {
    int i = blockIdx.x * blockDim.x + threadIdx.x;
    if (i < n) out[i] = f2bf(in[i]);
}

__global__ void k_fill_f32(float* __restrict__ p, float v, int n) {
    int i = blockIdx.x * blockDim.x + threadIdx.x;
    if (i < n) p[i] = v;
}

__global__ void k_fill_i32(int* __restrict__ p, int v, int n) {
    int i = blockIdx.x * blockDim.x + threadIdx.x;
    if (i < n) p[i] = v;
}

// ---------------------------------------------------------------------------
// C[M,Nout] = A[M,K] * B[Nout,K]^T    (A,B bf16 row-major, C f32)
// One wave computes a 16(M) x 64(N) block: 4 independent WMMA chains.
// Per-lane fragments: contiguous k = 16*hi + s in both A and B (16-bit A 16x32, B 32x16).
// One-stage software pipeline with __builtin_amdgcn_sched_barrier(0) between
// the next-iteration load block and the current WMMA chain: the backend may
// not sink the loads past the barrier, so waits before the WMMAs only cover
// the *previous* iteration's loads while 10 b128 loads remain in flight.
// NOTE: the one-step overfetch past k=K stays inside the workspace for all
// operands (each operand region is followed by more workspace), fault-safe.
// ---------------------------------------------------------------------------
template <int K>
__global__ void __launch_bounds__(256)
k_gemm_bf16(const unsigned short* __restrict__ A,
            const unsigned short* __restrict__ B,
            float* __restrict__ C, int M, int Nout) {
    int wave = (int)((blockIdx.x * blockDim.x + threadIdx.x) >> 5);
    int lane = threadIdx.x & 31;
    int tilesN = Nout >> 6;                  // 64-wide N blocks
    int total  = (M >> 4) * tilesN;
    if (wave >= total) return;               // wave-uniform: EXEC stays all-1s
    int m0  = (wave / tilesN) << 4;
    int n0  = (wave % tilesN) << 6;
    int low = lane & 15;
    int hi  = lane >> 4;

    const unsigned short* arow = A + (size_t)(m0 + low) * K + 16 * hi;
    const unsigned short* brow = B + (size_t)(n0 + low) * K + 16 * hi;

    v8f acc[4] = {v8f{}, v8f{}, v8f{}, v8f{}};

    // ---- prologue: fragments for k = 0 ------------------------------------
    v8bf alo = *reinterpret_cast<const v8bf*>(arow);
    v8bf ahi = *reinterpret_cast<const v8bf*>(arow + 8);
    v16bf a  = __builtin_shufflevector(alo, ahi,
                 0,1,2,3,4,5,6,7,8,9,10,11,12,13,14,15);
    v16bf b0 = *reinterpret_cast<const v16bf*>(brow);
    v16bf b1 = *reinterpret_cast<const v16bf*>(brow + (size_t)16 * K);
    v16bf b2 = *reinterpret_cast<const v16bf*>(brow + (size_t)32 * K);
    v16bf b3 = *reinterpret_cast<const v16bf*>(brow + (size_t)48 * K);

    #pragma unroll 1
    for (int k = 0; k < K; k += 32) {
        // issue next iteration's loads first (kept in flight across WMMAs)
        int kn = k + 32;                      // safe overfetch on last step
        __builtin_prefetch((const void*)(arow + kn + 224), 0, 1);
        v8bf nalo = *reinterpret_cast<const v8bf*>(arow + kn);
        v8bf nahi = *reinterpret_cast<const v8bf*>(arow + kn + 8);
        v16bf nb0 = *reinterpret_cast<const v16bf*>(brow + kn);
        v16bf nb1 = *reinterpret_cast<const v16bf*>(brow + (size_t)16 * K + kn);
        v16bf nb2 = *reinterpret_cast<const v16bf*>(brow + (size_t)32 * K + kn);
        v16bf nb3 = *reinterpret_cast<const v16bf*>(brow + (size_t)48 * K + kn);

        // hard scheduling fence: loads above may not sink below this point,
        // WMMAs below may not hoist above it.
        __builtin_amdgcn_sched_barrier(0);

        // 4 independent accumulator chains on current fragments
        acc[0] = __builtin_amdgcn_wmma_f32_16x16x32_bf16(false, a, false, b0,
                    (short)0, acc[0], false, false);
        acc[1] = __builtin_amdgcn_wmma_f32_16x16x32_bf16(false, a, false, b1,
                    (short)0, acc[1], false, false);
        acc[2] = __builtin_amdgcn_wmma_f32_16x16x32_bf16(false, a, false, b2,
                    (short)0, acc[2], false, false);
        acc[3] = __builtin_amdgcn_wmma_f32_16x16x32_bf16(false, a, false, b3,
                    (short)0, acc[3], false, false);

        a  = __builtin_shufflevector(nalo, nahi,
               0,1,2,3,4,5,6,7,8,9,10,11,12,13,14,15);
        b0 = nb0; b1 = nb1; b2 = nb2; b3 = nb3;
    }

    // C/D layout: VGPR r -> row m0 + r + 8*hi, col n0 + 16*t + low
    #pragma unroll
    for (int t = 0; t < 4; ++t) {
        float* crow = C + (size_t)(m0 + 8 * hi) * Nout + n0 + 16 * t + low;
        #pragma unroll
        for (int r = 0; r < 8; ++r)
            crow[(size_t)r * Nout] = acc[t][r];
    }
}

// ---------------------------------------------------------------------------
// a_src[n,h] = <h[n,h,:], att_src[h,:]> ; same for a_dst. 1 wave per (n,h).
__global__ void k_att(const float* __restrict__ h,
                      const float* __restrict__ attS,
                      const float* __restrict__ attD,
                      float* __restrict__ aS, float* __restrict__ aD,
                      int nNodes, int heads, int ch) {
    int wid  = (int)((blockIdx.x * blockDim.x + threadIdx.x) >> 5);
    int lane = threadIdx.x & 31;
    if (wid >= nNodes * heads) return;
    int n = wid / heads, hd = wid % heads;
    const float* hp = h + (size_t)n * heads * ch + (size_t)hd * ch;
    const float* sp = attS + (size_t)hd * ch;
    const float* dp = attD + (size_t)hd * ch;
    float s = 0.f, d = 0.f;
    for (int c = lane; c < ch; c += 32) {
        float v = hp[c];
        s += v * sp[c];
        d += v * dp[c];
    }
    #pragma unroll
    for (int off = 16; off > 0; off >>= 1) {
        s += __shfl_xor(s, off, 32);
        d += __shfl_xor(d, off, 32);
    }
    if (lane == 0) { aS[wid] = s; aD[wid] = d; }
}

// e = leaky_relu(aS[src]+aD[dst]); segment max via mapped-int atomicMax.
__global__ void k_edge_score(const int* __restrict__ src,
                             const int* __restrict__ dst,
                             const float* __restrict__ aS,
                             const float* __restrict__ aD,
                             float* __restrict__ ebuf, int* __restrict__ mbuf,
                             int E, int heads) {
    int i = blockIdx.x * blockDim.x + threadIdx.x;
    if (i >= E * heads) return;
    int e = i / heads, hd = i % heads;
    int s = src[e], d = dst[e];
    float v = aS[s * heads + hd] + aD[d * heads + hd];
    v = v > 0.f ? v : v * NEGSLOPE;
    ebuf[i] = v;
    atomicMax(&mbuf[d * heads + hd], fmap(v));
}

// ex = exp(e - max[dst]); denom[dst] += ex
__global__ void k_edge_exp(const int* __restrict__ dst,
                           float* __restrict__ ebuf,
                           const int* __restrict__ mbuf,
                           float* __restrict__ denom, int E, int heads) {
    int i = blockIdx.x * blockDim.x + threadIdx.x;
    if (i >= E * heads) return;
    int e = i / heads, hd = i % heads;
    int d = dst[e];
    float v = __expf(ebuf[i] - funmap(mbuf[d * heads + hd]));
    ebuf[i] = v;
    atomicAdd(&denom[d * heads + hd], v);
}

// agg[dst,h,:] += alpha * h[src,h,:]   (1 wave per (edge,head), coalesced)
__global__ void k_scatter(const int* __restrict__ src,
                          const int* __restrict__ dst,
                          const float* __restrict__ ebuf,
                          const float* __restrict__ denom,
                          const float* __restrict__ h,
                          float* __restrict__ agg,
                          int E, int heads, int ch) {
    int wid  = (int)((blockIdx.x * blockDim.x + threadIdx.x) >> 5);
    int lane = threadIdx.x & 31;
    if (wid >= E * heads) return;
    int e = wid / heads, hd = wid % heads;
    int s = src[e], d = dst[e];
    float alpha = ebuf[wid] / (denom[d * heads + hd] + EPSF);
    const float* hp = h   + (size_t)s * heads * ch + (size_t)hd * ch;
    float*       op = agg + (size_t)d * heads * ch + (size_t)hd * ch;
    for (int c = lane; c < ch; c += 32)
        atomicAdd(&op[c], alpha * hp[c]);
}

// layer-1 epilogue: relu(agg + b1) -> bf16 input of GEMM2
__global__ void k_post1(const float* __restrict__ agg,
                        const float* __restrict__ b1,
                        unsigned short* __restrict__ h2b, int total, int dim) {
    int i = blockIdx.x * blockDim.x + threadIdx.x;
    if (i >= total) return;
    float v = agg[i] + b1[i % dim];
    h2b[i] = f2bf(v > 0.f ? v : 0.f);
}

__global__ void k_bias_inplace(float* __restrict__ out,
                               const float* __restrict__ b, int total, int dim) {
    int i = blockIdx.x * blockDim.x + threadIdx.x;
    if (i < total) out[i] += b[i % dim];
}

// ---------------------------------------------------------------------------
extern "C" void kernel_launch(void* const* d_in, const int* in_sizes, int n_in,
                              void* d_out, int out_size, void* d_ws, size_t ws_size,
                              hipStream_t stream) {
    (void)n_in; (void)out_size; (void)ws_size;
    const float* x     = (const float*)d_in[0];
    const int*   ei    = (const int*)  d_in[1];
    const float* W1    = (const float*)d_in[2];
    const float* attS1 = (const float*)d_in[3];
    const float* attD1 = (const float*)d_in[4];
    const float* b1    = (const float*)d_in[5];
    const float* W2    = (const float*)d_in[6];
    const float* attS2 = (const float*)d_in[7];
    const float* attD2 = (const float*)d_in[8];
    const float* b2    = (const float*)d_in[9];
    float* out = (float*)d_out;

    const int E = in_sizes[1] / 2;
    const int* src = ei;
    const int* dst = ei + E;

    char* ws = (char*)d_ws;
    unsigned short* xb    = (unsigned short*)(ws + OFF_XB);
    unsigned short* W1b   = (unsigned short*)(ws + OFF_W1B);
    unsigned short* W2b   = (unsigned short*)(ws + OFF_W2B);
    float*          h1    = (float*)(ws + OFF_H1);            // layer-1 features
    unsigned short* h2b   = (unsigned short*)(ws + OFF_H1);   // reuse after scatter1
    float*          aS    = (float*)(ws + OFF_ATT);
    float*          aD    = (float*)(ws + OFF_ATT + 320000);
    int*            mmax  = (int*)  (ws + OFF_ATT + 640000);
    float*          denom = (float*)(ws + OFF_ATT + 960000);
    float*          ebuf  = (float*)(ws + OFF_EX);
    float*          agg1  = (float*)(ws + OFF_AGG);
    float*          h2    = (float*)(ws + OFF_AGG);           // reuse after post1

    const int B = 256;
    #define NB(n) (((n) + B - 1) / B)

    // ---- bf16 converts ----------------------------------------------------
    k_f2bf<<<NB(NNODES * F_IN), B, 0, stream>>>(x,  xb,  NNODES * F_IN);
    k_f2bf<<<NB(DIM1 * F_IN),   B, 0, stream>>>(W1, W1b, DIM1 * F_IN);
    k_f2bf<<<NB(DIM2 * DIM1),   B, 0, stream>>>(W2, W2b, DIM2 * DIM1);

    // ---- layer 1 ----------------------------------------------------------
    {   // GEMM1: waves = (10000/16)*(1024/64) = 10000
        int waves = (NNODES / 16) * (DIM1 / 64);
        k_gemm_bf16<F_IN><<<NB(waves * 32), B, 0, stream>>>(xb, W1b, h1, NNODES, DIM1);
    }
    k_att<<<NB(NNODES * HEADS1 * 32), B, 0, stream>>>(h1, attS1, attD1, aS, aD,
                                                      NNODES, HEADS1, CH1);
    k_fill_i32<<<NB(NNODES * HEADS1), B, 0, stream>>>(mmax, (int)0x80000000, NNODES * HEADS1);
    k_fill_f32<<<NB(NNODES * HEADS1), B, 0, stream>>>(denom, 0.f, NNODES * HEADS1);
    k_fill_f32<<<NB(NNODES * DIM1),   B, 0, stream>>>(agg1, 0.f, NNODES * DIM1);

    k_edge_score<<<NB(E * HEADS1), B, 0, stream>>>(src, dst, aS, aD, ebuf, mmax, E, HEADS1);
    k_edge_exp  <<<NB(E * HEADS1), B, 0, stream>>>(dst, ebuf, mmax, denom, E, HEADS1);
    k_scatter   <<<NB(E * HEADS1 * 32), B, 0, stream>>>(src, dst, ebuf, denom, h1, agg1,
                                                        E, HEADS1, CH1);
    k_post1<<<NB(NNODES * DIM1), B, 0, stream>>>(agg1, b1, h2b, NNODES * DIM1, DIM1);

    // ---- layer 2 ----------------------------------------------------------
    {   // GEMM2: waves = (10000/16)*(768/64) = 7500
        int waves = (NNODES / 16) * (DIM2 / 64);
        k_gemm_bf16<DIM1><<<NB(waves * 32), B, 0, stream>>>(h2b, W2b, h2, NNODES, DIM2);
    }
    k_att<<<NB(NNODES * 32), B, 0, stream>>>(h2, attS2, attD2, aS, aD, NNODES, 1, DIM2);
    k_fill_i32<<<NB(NNODES), B, 0, stream>>>(mmax, (int)0x80000000, NNODES);
    k_fill_f32<<<NB(NNODES), B, 0, stream>>>(denom, 0.f, NNODES);
    k_fill_f32<<<NB(NNODES * DIM2), B, 0, stream>>>(out, 0.f, NNODES * DIM2);

    k_edge_score<<<NB(E), B, 0, stream>>>(src, dst, aS, aD, ebuf, mmax, E, 1);
    k_edge_exp  <<<NB(E), B, 0, stream>>>(dst, ebuf, mmax, denom, E, 1);
    k_scatter   <<<NB(E * 32), B, 0, stream>>>(src, dst, ebuf, denom, h2, out, E, 1, DIM2);
    k_bias_inplace<<<NB(NNODES * DIM2), B, 0, stream>>>(out, b2, NNODES * DIM2, DIM2);
    #undef NB
}
